// DARNN_32134945308819
// MI455X (gfx1250) — matrix operand order repaired and checked by reading, not compiled, not run on hardware
//
#include <hip/hip_runtime.h>
#include <hip/hip_bf16.h>

typedef __attribute__((ext_vector_type(16))) _Float16 v16h;
typedef __attribute__((ext_vector_type(8)))  _Float16 v8h;
typedef __attribute__((ext_vector_type(8)))  float    v8f;

#define B_TOT 512
#define T_LEN 64
#define I_DIM 128
#define E_DIM 127
#define H_DIM 128
#define G_DIM 512   // 4*H
#define MB    16    // batch rows per workgroup

// ---------------------------------------------------------------------------
// WMMA helpers (CDNA5 v_wmma_f32_16x16x32_f16, wave32)
// ---------------------------------------------------------------------------
__device__ __forceinline__ v8f wmma16(v16h a, v16h b, v8f c) {
  return __builtin_amdgcn_wmma_f32_16x16x32_f16(false, a, false, b, (short)0, c,
                                                false, false);
}

// A-matrix 16x32 f16 fragment. lane m=lane&15, grp=lane>>4;
// half p -> K = grp*16 + p  => one contiguous 32B chunk (matches B).
__device__ __forceinline__ v16h load_a_f16(const _Float16* mat, int stride, int k0) {
  int lane = threadIdx.x & 31;
  int m = lane & 15, grp = lane >> 4;
  const _Float16* p = mat + (size_t)m * stride + k0 + grp * 16;
  return *(const v16h*)(p);
}

// B-matrix 32x16 f16 fragment: computes X @ W^T with W row-major (N x K);
// lane n reads 16 contiguous halves of W row n at K = k0 + grp*16 (32B load).
__device__ __forceinline__ v16h load_b_f16(const _Float16* mat, int stride, int n0, int k0) {
  int lane = threadIdx.x & 31;
  int n = n0 + (lane & 15), grp = lane >> 4;
  return *(const v16h*)(mat + (size_t)n * stride + k0 + grp * 16);
}

// D 16x16 f32: lane (n=lane&15, grp), VGPR r -> M = r + 8*grp
__device__ __forceinline__ void store_d_f32(float* mat, int stride, int n0, v8f d) {
  int lane = threadIdx.x & 31;
  int n = n0 + (lane & 15), grp = lane >> 4;
#pragma unroll
  for (int r = 0; r < 8; ++r) mat[(size_t)(r + 8 * grp) * stride + n] = d[r];
}

__device__ __forceinline__ void store_d_f16(_Float16* mat, int stride, int n0, v8f d) {
  int lane = threadIdx.x & 31;
  int n = n0 + (lane & 15), grp = lane >> 4;
#pragma unroll
  for (int r = 0; r < 8; ++r) mat[(size_t)(r + 8 * grp) * stride + n] = (_Float16)d[r];
}

__device__ __forceinline__ float sigf(float x) { return 1.0f / (1.0f + expf(-x)); }

// ---------------------------------------------------------------------------
// Tensor Data Mover: one-shot 2D tile (rows x dim0 halves, packed) -> LDS.
// D# per CDNA5 ISA. 6-arg builtin form (clang-23).
// ---------------------------------------------------------------------------
#if __has_builtin(__builtin_amdgcn_tensor_load_to_lds) && \
    __has_builtin(__builtin_amdgcn_s_wait_tensorcnt)
#define HAVE_TDM 1
typedef __attribute__((ext_vector_type(4))) unsigned int u32x4;
typedef __attribute__((ext_vector_type(8))) int i32x8;
typedef __attribute__((ext_vector_type(4))) int i32x4;

__device__ __forceinline__ void tdm_load_2d_f16(unsigned lds_off, const _Float16* gsrc,
                                                unsigned dim0, unsigned rows) {
  unsigned long long ga = (unsigned long long)(uintptr_t)gsrc;
  u32x4 g0;
  g0.x = 1u;                                                // count=1, user desc
  g0.y = lds_off;                                           // lds_addr
  g0.z = (unsigned)(ga & 0xFFFFFFFFu);                      // global_addr[31:0]
  g0.w = (unsigned)((ga >> 32) & 0x1FFFFFFu) | (2u << 30);  // addr[56:32] | type=2
  i32x8 g1;
  g1[0] = (int)(1u << 16);                                  // data_size=1 (2 bytes)
  g1[1] = (int)((dim0 & 0xFFFFu) << 16);                    // tensor_dim0[15:0]
  g1[2] = (int)(((dim0 >> 16) & 0xFFFFu) | ((rows & 0xFFFFu) << 16)); // dim0 hi|dim1 lo
  g1[3] = (int)(((rows >> 16) & 0xFFFFu) | ((dim0 & 0xFFFFu) << 16)); // dim1 hi|tile_dim0
  g1[4] = (int)(rows & 0xFFFFu);                            // tile_dim1 (tile_dim2=0)
  g1[5] = (int)dim0;                                        // tensor_dim0_stride lo32
  g1[6] = 0;
  g1[7] = 0;
  i32x4 z4 = {0, 0, 0, 0};
  i32x8 z8 = {0, 0, 0, 0, 0, 0, 0, 0};
  __builtin_amdgcn_tensor_load_to_lds(g0, g1, z4, z4, z8, 0);
}
#endif

// ---------------------------------------------------------------------------
// Prep: f32 -> f16 weight conversion with zero-padding / column slice
// ---------------------------------------------------------------------------
__global__ void conv_f16_kernel(const float* __restrict__ src, _Float16* __restrict__ dst,
                                int rows, int cols_out, int src_stride, int src_off,
                                int src_cols) {
  int gid = blockIdx.x * 256 + threadIdx.x;
  int total = rows * cols_out;
  if (gid >= total) return;
  int r = gid / cols_out, c = gid - r * cols_out;
  float v = (c < src_cols) ? src[(size_t)r * src_stride + src_off + c] : 0.0f;
  dst[gid] = (_Float16)v;
}

// ---------------------------------------------------------------------------
// Prep: P[b][e][t] = sum_k input[b][k][1+e] * attn_W1[t][256+k]  (time-invariant)
// ---------------------------------------------------------------------------
__global__ void p_kernel(const float* __restrict__ input, const float* __restrict__ attW1,
                         _Float16* __restrict__ P) {
  size_t gid = (size_t)blockIdx.x * 256 + threadIdx.x;
  if (gid >= (size_t)B_TOT * E_DIM * T_LEN) return;
  int t = (int)(gid & 63);
  size_t r = gid >> 6;
  int e = (int)(r % E_DIM);
  int b = (int)(r / E_DIM);
  const float* w = attW1 + (size_t)t * 320 + 256;
  const float* x = input + (size_t)b * T_LEN * I_DIM + 1 + e;
  float acc = 0.0f;
#pragma unroll 8
  for (int k = 0; k < 64; ++k) acc += x[(size_t)k * I_DIM] * w[k];
  P[gid] = (_Float16)acc;
}

// ---------------------------------------------------------------------------
// Prep: Q = h_exp @ attnd_W1[:,256:384]^T  (M = B*T, N = K = 128), WMMA -> f16
// ---------------------------------------------------------------------------
__global__ __launch_bounds__(256) void q_kernel(const _Float16* __restrict__ hexp,
                                                const _Float16* __restrict__ We,
                                                _Float16* __restrict__ Q) {
  int wid = threadIdx.x >> 5;           // 8 waves -> 8 N-tiles
  size_t m0 = (size_t)blockIdx.x * 16;  // 2048 blocks cover 32768 rows
  const _Float16* A = hexp + m0 * 128;
  v8f acc = {};
#pragma unroll
  for (int kc = 0; kc < 4; ++kc)
    acc = wmma16(load_a_f16(A, 128, kc * 32), load_b_f16(We, 128, wid * 16, kc * 32), acc);
  store_d_f16(Q + m0 * 128, 128, wid * 16, acc);
}

// ---------------------------------------------------------------------------
// Encoder: input attention + 2-layer LSTM, sequential over T. 1 WG = 16 batches.
// Dynamic LDS: sP = staged P slice (MB*E_DIM*T_LEN f16 = 254 KB; WGP has 320 KB).
// ---------------------------------------------------------------------------
__global__ __launch_bounds__(256) void darnn_encoder(
    const float* __restrict__ input, const _Float16* __restrict__ Wih0,
    const _Float16* __restrict__ Whh0, const _Float16* __restrict__ Wih1,
    const _Float16* __restrict__ Whh1, const _Float16* __restrict__ Watt,
    const _Float16* __restrict__ Pf, const float* __restrict__ ab1,
    const float* __restrict__ aw2, const float* __restrict__ ab2,
    const float* __restrict__ bih0, const float* __restrict__ bhh0,
    const float* __restrict__ bih1, const float* __restrict__ bhh1,
    _Float16* __restrict__ hexp) {
  __shared__ __align__(16) float sGates[MB * G_DIM];   // reused: S[0..1024) logits[1024..3072)
  __shared__ __align__(32) _Float16 sXS[MB * 128];
  __shared__ __align__(32) _Float16 sH0[MB * 128];
  __shared__ __align__(32) _Float16 sH1[MB * 128];
  __shared__ __align__(32) _Float16 sHC[MB * 256];     // [h1 | c1] f16
  __shared__ float sB1[64], sW2[64], sBias0[512], sBias1[512];
  extern __shared__ _Float16 sP[];                     // MB * E_DIM * T_LEN

  const int tid = threadIdx.x;
  const int wid = tid >> 5;
  const int bb0 = blockIdx.x * MB;
  const _Float16* psrc = Pf + (size_t)bb0 * E_DIM * T_LEN;

#if defined(HAVE_TDM)
  if (wid == 0)  // TDM DMA: whole time-invariant P slice -> LDS (overlaps init)
    tdm_load_2d_f16(__builtin_amdgcn_groupstaticsize(), psrc,
                    (unsigned)(E_DIM * T_LEN), (unsigned)MB);
#endif

  for (int i = tid; i < MB * 128; i += 256) { sH0[i] = (_Float16)0.f; sH1[i] = (_Float16)0.f; }
  for (int i = tid; i < MB * 256; i += 256) sHC[i] = (_Float16)0.f;
  if (tid < 64) { sB1[tid] = ab1[tid]; sW2[tid] = aw2[tid]; }
  for (int i = tid; i < 512; i += 256) { sBias0[i] = bih0[i] + bhh0[i]; sBias1[i] = bih1[i] + bhh1[i]; }
  float c0r[8], c1r[8];
#pragma unroll
  for (int u = 0; u < 8; ++u) { c0r[u] = 0.f; c1r[u] = 0.f; }
  const float b2v = ab2[0];
  // Warm weight streams into cache for the first step.
  __builtin_prefetch(Wih0 + tid * 64, 0, 0);
  __builtin_prefetch(Whh0 + tid * 64, 0, 0);

#if defined(HAVE_TDM)
  if (wid == 0) __builtin_amdgcn_s_wait_tensorcnt(0);
#else
  for (int i = tid; i < MB * E_DIM * T_LEN; i += 256) sP[i] = psrc[i];
#endif
  __syncthreads();

  float* const sS   = sGates;            // 16 x 64
  float* const sLog = sGates + MB * 64;  // 16 x 128 (127 used)

  for (int t = 0; t < T_LEN; ++t) {
    // ---- attention scores S = [h1|c1](16x256) @ Watt^T(256x64) ----
    if (wid < 4) {
      v8f acc = {};
#pragma unroll
      for (int kc = 0; kc < 8; ++kc)
        acc = wmma16(load_a_f16(sHC, 256, kc * 32),
                     load_b_f16(Watt, 256, wid * 16, kc * 32), acc);
      store_d_f32(sS, 64, wid * 16, acc);
    }
    __syncthreads();
    // ---- logits[b][e] = sum_t2 tanh(P + S + b1) * w2  (P from LDS) ----
    for (int pi = tid; pi < MB * E_DIM; pi += 256) {
      int b = pi / E_DIM, e = pi - b * E_DIM;
      const _Float16* pr = sP + ((size_t)b * E_DIM + e) * 64;
      float acc = b2v;
#pragma unroll 8
      for (int t2 = 0; t2 < 64; ++t2) {
        float v = tanhf((float)pr[t2] + sS[b * 64 + t2] + sB1[t2]);
        acc += v * sW2[t2];
      }
      sLog[b * 128 + e] = acc;
    }
    __syncthreads();
    // ---- softmax over e ----
    if (tid < MB) {
      int b = tid;
      float mx = -1e30f;
      for (int e = 0; e < E_DIM; ++e) mx = fmaxf(mx, sLog[b * 128 + e]);
      float sum = 0.f;
      for (int e = 0; e < E_DIM; ++e) { float ex = expf(sLog[b*128+e] - mx); sLog[b*128+e] = ex; sum += ex; }
      float inv = 1.f / sum;
      for (int e = 0; e < E_DIM; ++e) sLog[b * 128 + e] *= inv;
      sXS[b * 128 + 127] = (_Float16)0.f;  // K-pad column
    }
    __syncthreads();
    // ---- weighted input xs = x_t * a ----
    for (int pi = tid; pi < MB * E_DIM; pi += 256) {
      int b = pi / E_DIM, e = pi - b * E_DIM;
      float x = input[((size_t)(bb0 + b) * T_LEN + t) * I_DIM + 1 + e];
      sXS[b * 128 + e] = (_Float16)(x * sLog[b * 128 + e]);
    }
    __syncthreads();
    // ---- cell0 gates: xs@Wih0^T + h0@Whh0^T (16x512) ----
#pragma unroll
    for (int q = 0; q < 4; ++q) {
      int n0 = (wid * 4 + q) * 16;
      v8f acc = {};
#pragma unroll
      for (int kc = 0; kc < 4; ++kc) {
        acc = wmma16(load_a_f16(sXS, 128, kc * 32), load_b_f16(Wih0, 128, n0, kc * 32), acc);
        acc = wmma16(load_a_f16(sH0, 128, kc * 32), load_b_f16(Whh0, 128, n0, kc * 32), acc);
      }
      store_d_f32(sGates, G_DIM, n0, acc);
    }
    __syncthreads();
    // ---- cell0 elementwise ----
#pragma unroll
    for (int u = 0; u < 8; ++u) {
      int idx = tid + 256 * u;
      int b = idx >> 7, j = idx & 127;
      float ig = sGates[b * 512 + j]       + sBias0[j];
      float fg = sGates[b * 512 + 128 + j] + sBias0[128 + j];
      float gg = sGates[b * 512 + 256 + j] + sBias0[256 + j];
      float og = sGates[b * 512 + 384 + j] + sBias0[384 + j];
      float c = sigf(fg) * c0r[u] + sigf(ig) * tanhf(gg);
      c0r[u] = c;
      sH0[b * 128 + j] = (_Float16)(sigf(og) * tanhf(c));
    }
    __syncthreads();
    // ---- cell1 gates: h0@Wih1^T + h1@Whh1^T ----
#pragma unroll
    for (int q = 0; q < 4; ++q) {
      int n0 = (wid * 4 + q) * 16;
      v8f acc = {};
#pragma unroll
      for (int kc = 0; kc < 4; ++kc) {
        acc = wmma16(load_a_f16(sH0, 128, kc * 32), load_b_f16(Wih1, 128, n0, kc * 32), acc);
        acc = wmma16(load_a_f16(sH1, 128, kc * 32), load_b_f16(Whh1, 128, n0, kc * 32), acc);
      }
      store_d_f32(sGates, G_DIM, n0, acc);
    }
    __syncthreads();
    // ---- cell1 elementwise + emit h_exp ----
#pragma unroll
    for (int u = 0; u < 8; ++u) {
      int idx = tid + 256 * u;
      int b = idx >> 7, j = idx & 127;
      float ig = sGates[b * 512 + j]       + sBias1[j];
      float fg = sGates[b * 512 + 128 + j] + sBias1[128 + j];
      float gg = sGates[b * 512 + 256 + j] + sBias1[256 + j];
      float og = sGates[b * 512 + 384 + j] + sBias1[384 + j];
      float c = sigf(fg) * c1r[u] + sigf(ig) * tanhf(gg);
      c1r[u] = c;
      float h = sigf(og) * tanhf(c);
      _Float16 hh = (_Float16)h;
      sH1[b * 128 + j] = hh;
      sHC[b * 256 + j] = hh;
      sHC[b * 256 + 128 + j] = (_Float16)c;
      hexp[((size_t)(bb0 + b) * T_LEN + t) * 128 + j] = hh;
    }
    __syncthreads();
  }
}

// ---------------------------------------------------------------------------
// Decoder: temporal attention + 2-layer LSTM + projection.
// Dynamic LDS: sQ = staged Q slice (MB*T_LEN*H f16 = 256 KB).
// ---------------------------------------------------------------------------
__global__ __launch_bounds__(256) void darnn_decoder(
    const float* __restrict__ input, const _Float16* __restrict__ Whh0,
    const _Float16* __restrict__ Wih1, const _Float16* __restrict__ Whh1,
    const _Float16* __restrict__ Wattd, const _Float16* __restrict__ hexp,
    const _Float16* __restrict__ Q, const float* __restrict__ adb1,
    const float* __restrict__ adw2, const float* __restrict__ adb2,
    const float* __restrict__ wih0v, const float* __restrict__ bih0,
    const float* __restrict__ bhh0, const float* __restrict__ bih1,
    const float* __restrict__ bhh1, const float* __restrict__ decinW,
    const float* __restrict__ decinB, const float* __restrict__ projW,
    const float* __restrict__ projB, float* __restrict__ out) {
  __shared__ __align__(16) float sGates[MB * G_DIM];  // SD[0..2048) logD[2048..3072) att[3072..4096)
  __shared__ float sPartial[MB * 128];
  __shared__ __align__(32) _Float16 sH0[MB * 128];
  __shared__ __align__(32) _Float16 sH1[MB * 128];
  __shared__ __align__(32) _Float16 sHC[MB * 256];
  __shared__ float sDin[MB];
  __shared__ float sDb0[512], sDb1[512], sAb1[128], sAw2[128];
  extern __shared__ _Float16 sQ[];                    // MB * T_LEN * H_DIM

  const int tid = threadIdx.x;
  const int wid = tid >> 5;
  const int bb0 = blockIdx.x * MB;
  const _Float16* qsrc = Q + (size_t)bb0 * T_LEN * H_DIM;

#if defined(HAVE_TDM)
  if (wid == 0)
    tdm_load_2d_f16(__builtin_amdgcn_groupstaticsize(), qsrc,
                    (unsigned)(T_LEN * H_DIM), (unsigned)MB);
#endif

  for (int i = tid; i < MB * 128; i += 256) { sH0[i] = (_Float16)0.f; sH1[i] = (_Float16)0.f; }
  for (int i = tid; i < MB * 256; i += 256) sHC[i] = (_Float16)0.f;
  for (int i = tid; i < 512; i += 256) { sDb0[i] = bih0[i] + bhh0[i]; sDb1[i] = bih1[i] + bhh1[i]; }
  if (tid < 128) { sAb1[tid] = adb1[tid]; sAw2[tid] = adw2[tid]; }
  float c0r[8], c1r[8];
#pragma unroll
  for (int u = 0; u < 8; ++u) { c0r[u] = 0.f; c1r[u] = 0.f; }
  const float b2v = adb2[0];
  __builtin_prefetch(Whh0 + tid * 64, 0, 0);
  __builtin_prefetch(Wih1 + tid * 64, 0, 0);

#if defined(HAVE_TDM)
  if (wid == 0) __builtin_amdgcn_s_wait_tensorcnt(0);
#else
  for (int i = tid; i < MB * T_LEN * H_DIM; i += 256) sQ[i] = qsrc[i];
#endif
  __syncthreads();

  float* const sSD  = sGates;
  float* const sLg  = sGates + 2048;
  float* const sAtt = sGates + 3072;

  for (int t = 0; t < T_LEN; ++t) {
    // ---- SD = [h1|c1](16x256) @ Wattd^T(256x128) ----
    {
      v8f acc = {};
#pragma unroll
      for (int kc = 0; kc < 8; ++kc)
        acc = wmma16(load_a_f16(sHC, 256, kc * 32),
                     load_b_f16(Wattd, 256, wid * 16, kc * 32), acc);
      store_d_f32(sSD, 128, wid * 16, acc);
    }
    __syncthreads();
    // ---- logits[b][t2] = sum_j tanh(Q + SD + b1)*w2  (Q from LDS) ----
    for (int pi = tid; pi < MB * T_LEN; pi += 256) {
      int b = pi >> 6, t2 = pi & 63;
      const _Float16* qr = sQ + ((size_t)b * T_LEN + t2) * 128;
      float acc = b2v;
#pragma unroll 8
      for (int j = 0; j < 128; ++j) {
        float v = tanhf((float)qr[j] + sSD[b * 128 + j] + sAb1[j]);
        acc += v * sAw2[j];
      }
      sLg[b * 64 + t2] = acc;
    }
    __syncthreads();
    // ---- softmax over t2 ----
    if (tid < MB) {
      int b = tid;
      float mx = -1e30f;
      for (int k = 0; k < 64; ++k) mx = fmaxf(mx, sLg[b * 64 + k]);
      float sum = 0.f;
      for (int k = 0; k < 64; ++k) { float ex = expf(sLg[b*64+k] - mx); sAtt[b*64+k] = ex; sum += ex; }
      float inv = 1.f / sum;
      for (int k = 0; k < 64; ++k) sAtt[b * 64 + k] *= inv;
    }
    __syncthreads();
    // ---- context: partial[b][j] = sum_t2 a[b][t2] * hexp[b][t2][j] ----
#pragma unroll
    for (int u = 0; u < 8; ++u) {
      int idx = tid + 256 * u;
      int b = idx >> 7, j = idx & 127;
      float acc = 0.f;
      const _Float16* hb = hexp + (size_t)(bb0 + b) * T_LEN * 128 + j;
#pragma unroll 8
      for (int k = 0; k < 64; ++k) acc += sAtt[b * 64 + k] * (float)hb[(size_t)k * 128];
      sPartial[b * 128 + j] = acc;
    }
    __syncthreads();
    // ---- decoder scalar input ----
    if (tid < MB) {
      int b = tid;
      float acc = decinB[0];
      for (int j = 0; j < 128; ++j) acc += sPartial[b * 128 + j] * decinW[j];
      acc += input[((size_t)(bb0 + b) * T_LEN + t) * I_DIM] * decinW[128];
      sDin[b] = acc;
    }
    __syncthreads();
    // ---- cell0 gates: recurrent h0@Whh0^T (din rank-1 term added below) ----
#pragma unroll
    for (int q = 0; q < 4; ++q) {
      int n0 = (wid * 4 + q) * 16;
      v8f acc = {};
#pragma unroll
      for (int kc = 0; kc < 4; ++kc)
        acc = wmma16(load_a_f16(sH0, 128, kc * 32), load_b_f16(Whh0, 128, n0, kc * 32), acc);
      store_d_f32(sGates, G_DIM, n0, acc);
    }
    __syncthreads();
    // ---- cell0 elementwise ----
#pragma unroll
    for (int u = 0; u < 8; ++u) {
      int idx = tid + 256 * u;
      int b = idx >> 7, j = idx & 127;
      float din = sDin[b];
      float ig = sGates[b * 512 + j]       + din * wih0v[j]       + sDb0[j];
      float fg = sGates[b * 512 + 128 + j] + din * wih0v[128 + j] + sDb0[128 + j];
      float gg = sGates[b * 512 + 256 + j] + din * wih0v[256 + j] + sDb0[256 + j];
      float og = sGates[b * 512 + 384 + j] + din * wih0v[384 + j] + sDb0[384 + j];
      float c = sigf(fg) * c0r[u] + sigf(ig) * tanhf(gg);
      c0r[u] = c;
      sH0[b * 128 + j] = (_Float16)(sigf(og) * tanhf(c));
    }
    __syncthreads();
    // ---- cell1 gates ----
#pragma unroll
    for (int q = 0; q < 4; ++q) {
      int n0 = (wid * 4 + q) * 16;
      v8f acc = {};
#pragma unroll
      for (int kc = 0; kc < 4; ++kc) {
        acc = wmma16(load_a_f16(sH0, 128, kc * 32), load_b_f16(Wih1, 128, n0, kc * 32), acc);
        acc = wmma16(load_a_f16(sH1, 128, kc * 32), load_b_f16(Whh1, 128, n0, kc * 32), acc);
      }
      store_d_f32(sGates, G_DIM, n0, acc);
    }
    __syncthreads();
    // ---- cell1 elementwise ----
#pragma unroll
    for (int u = 0; u < 8; ++u) {
      int idx = tid + 256 * u;
      int b = idx >> 7, j = idx & 127;
      float ig = sGates[b * 512 + j]       + sDb1[j];
      float fg = sGates[b * 512 + 128 + j] + sDb1[128 + j];
      float gg = sGates[b * 512 + 256 + j] + sDb1[256 + j];
      float og = sGates[b * 512 + 384 + j] + sDb1[384 + j];
      float c = sigf(fg) * c1r[u] + sigf(ig) * tanhf(gg);
      c1r[u] = c;
      float h = sigf(og) * tanhf(c);
      _Float16 hh = (_Float16)h;
      sH1[b * 128 + j] = hh;
      sHC[b * 256 + j] = hh;
      sHC[b * 256 + 128 + j] = (_Float16)c;
    }
    __syncthreads();
  }
  // ---- final projection: out = [h1|partial] @ projW^T + projB ----
  if (tid < MB) {
    int b = tid;
    float acc = projB[0];
    for (int j = 0; j < 128; ++j)
      acc += (float)sH1[b * 128 + j] * projW[j] + sPartial[b * 128 + j] * projW[128 + j];
    out[bb0 + b] = acc;
  }
}

// ---------------------------------------------------------------------------
// Workspace layout (bytes, 256-aligned)
// ---------------------------------------------------------------------------
#define WS_EIH0  0u          // 512x128 f16
#define WS_EHH0  131072u
#define WS_EIH1  262144u
#define WS_EHH1  393216u
#define WS_DHH0  524288u
#define WS_DIH1  655360u
#define WS_DHH1  786432u
#define WS_ATT   917504u     // 64x256 f16
#define WS_ATTD  950272u     // 128x256 f16
#define WS_ATTE  1015808u    // 128x128 f16
#define WS_P     1048576u    // 512x127x64 f16
#define WS_HEXP  9371648u    // 512x64x128 f16
#define WS_Q     17760256u   // 512x64x128 f16

extern "C" void kernel_launch(void* const* d_in, const int* in_sizes, int n_in,
                              void* d_out, int out_size, void* d_ws, size_t ws_size,
                              hipStream_t stream) {
  (void)in_sizes; (void)n_in; (void)out_size; (void)ws_size;
  const float* input = (const float*)d_in[0];
  char* ws = (char*)d_ws;
  _Float16* wEih0 = (_Float16*)(ws + WS_EIH0);
  _Float16* wEhh0 = (_Float16*)(ws + WS_EHH0);
  _Float16* wEih1 = (_Float16*)(ws + WS_EIH1);
  _Float16* wEhh1 = (_Float16*)(ws + WS_EHH1);
  _Float16* wDhh0 = (_Float16*)(ws + WS_DHH0);
  _Float16* wDih1 = (_Float16*)(ws + WS_DIH1);
  _Float16* wDhh1 = (_Float16*)(ws + WS_DHH1);
  _Float16* wAtt  = (_Float16*)(ws + WS_ATT);
  _Float16* wAttd = (_Float16*)(ws + WS_ATTD);
  _Float16* wAtte = (_Float16*)(ws + WS_ATTE);
  _Float16* Pf    = (_Float16*)(ws + WS_P);
  _Float16* hexp  = (_Float16*)(ws + WS_HEXP);
  _Float16* Qh    = (_Float16*)(ws + WS_Q);

  dim3 blk(256);
  auto cdiv = [](long a, long b) { return (int)((a + b - 1) / b); };

  conv_f16_kernel<<<cdiv(512 * 128, 256), blk, 0, stream>>>((const float*)d_in[1],  wEih0, 512, 128, 127, 0, 127);
  conv_f16_kernel<<<cdiv(512 * 128, 256), blk, 0, stream>>>((const float*)d_in[2],  wEhh0, 512, 128, 128, 0, 128);
  conv_f16_kernel<<<cdiv(512 * 128, 256), blk, 0, stream>>>((const float*)d_in[5],  wEih1, 512, 128, 128, 0, 128);
  conv_f16_kernel<<<cdiv(512 * 128, 256), blk, 0, stream>>>((const float*)d_in[6],  wEhh1, 512, 128, 128, 0, 128);
  conv_f16_kernel<<<cdiv(512 * 128, 256), blk, 0, stream>>>((const float*)d_in[10], wDhh0, 512, 128, 128, 0, 128);
  conv_f16_kernel<<<cdiv(512 * 128, 256), blk, 0, stream>>>((const float*)d_in[13], wDih1, 512, 128, 128, 0, 128);
  conv_f16_kernel<<<cdiv(512 * 128, 256), blk, 0, stream>>>((const float*)d_in[14], wDhh1, 512, 128, 128, 0, 128);
  conv_f16_kernel<<<cdiv(64 * 256, 256),  blk, 0, stream>>>((const float*)d_in[17], wAtt,   64, 256, 320, 0, 256);
  conv_f16_kernel<<<cdiv(128 * 256, 256), blk, 0, stream>>>((const float*)d_in[21], wAttd, 128, 256, 384, 0, 256);
  conv_f16_kernel<<<cdiv(128 * 128, 256), blk, 0, stream>>>((const float*)d_in[21], wAtte, 128, 128, 384, 256, 128);

  p_kernel<<<cdiv((long)B_TOT * E_DIM * T_LEN, 256), blk, 0, stream>>>(
      input, (const float*)d_in[17], Pf);

  // encoder scan: 254 KB dynamic LDS for the staged P slice (WGP LDS = 320 KB)
  darnn_encoder<<<B_TOT / MB, blk, MB * E_DIM * T_LEN * sizeof(_Float16), stream>>>(
      input, wEih0, wEhh0, wEih1, wEhh1, wAtt, Pf,
      (const float*)d_in[18], (const float*)d_in[19], (const float*)d_in[20],
      (const float*)d_in[3], (const float*)d_in[4],
      (const float*)d_in[7], (const float*)d_in[8], hexp);

  q_kernel<<<(B_TOT * T_LEN) / 16, blk, 0, stream>>>(hexp, wAtte, Qh);

  // decoder scan: 256 KB dynamic LDS for the staged Q slice
  darnn_decoder<<<B_TOT / MB, blk, MB * T_LEN * H_DIM * sizeof(_Float16), stream>>>(
      input, wDhh0, wDih1, wDhh1, wAttd, hexp, Qh,
      (const float*)d_in[22], (const float*)d_in[23], (const float*)d_in[24],
      (const float*)d_in[9], (const float*)d_in[11], (const float*)d_in[12],
      (const float*)d_in[15], (const float*)d_in[16],
      (const float*)d_in[25], (const float*)d_in[26],
      (const float*)d_in[27], (const float*)d_in[28],
      (float*)d_out);
}
